// GPT2Attention_22213570855653
// MI455X (gfx1250) — compile-verified, matrix operand order checked by reading, not run on hardware
//
#include <hip/hip_runtime.h>
#include <hip/hip_bf16.h>

// ---------------------------------------------------------------------------
// GPT-2 attention block for MI455X (gfx1250, wave32, WMMA).
// All heavy math goes through v_wmma_f32_16x16x32_bf16 (fp32 accumulate).
// Global->LDS staging uses CDNA5 async-to-LDS DMA (ASYNCcnt) + double buffer.
// ---------------------------------------------------------------------------

typedef __attribute__((ext_vector_type(16))) __bf16 v16bf;
typedef __attribute__((ext_vector_type(8)))  float  v8f;

struct alignas(16) U4 { unsigned int x, y, z, w; };
union Frag { U4 u[2]; v16bf v; };

#define WMMA_BF16(A, B, C) \
    __builtin_amdgcn_wmma_f32_16x16x32_bf16(false, (A), false, (B), (short)0, (C), false, false)

// CDNA5 async global->LDS copy, 16B per lane, tracked by ASYNCcnt.
// VDST = LDS byte address (low 32 bits of the shared pointer), VADDR = 64-bit
// global address, no SADDR ("off").
__device__ __forceinline__ void async_copy16(__bf16* lds, const __bf16* g) {
    unsigned int l = (unsigned int)(unsigned long long)(uintptr_t)lds;
    asm volatile("global_load_async_to_lds_b128 %0, %1, off"
                 :: "v"(l), "v"((unsigned long long)(uintptr_t)g)
                 : "memory");
}
__device__ __forceinline__ void wait_async_le4() {
    asm volatile("s_wait_asynccnt 0x4" ::: "memory");
}
__device__ __forceinline__ void wait_async_le0() {
    asm volatile("s_wait_asynccnt 0x0" ::: "memory");
}

// ---------------------------------------------------------------------------
// Stage 0: fp32 -> bf16 conversions
// ---------------------------------------------------------------------------
__global__ void cvt_bf16_kernel(const float* __restrict__ in, __bf16* __restrict__ out, int n) {
    int i = blockIdx.x * blockDim.x + threadIdx.x;
    if (i < n) out[i] = (__bf16)in[i];
}

// in: [R][C] fp32 row-major -> out: [C][R] bf16 row-major (K-contiguous B for GEMM)
__global__ void cvt_transpose_bf16_kernel(const float* __restrict__ in, __bf16* __restrict__ out,
                                          int R, int C) {
    int i = blockIdx.x * blockDim.x + threadIdx.x;
    int n = R * C;
    if (i < n) {
        int c = i / R;
        int r = i - c * R;
        out[i] = (__bf16)in[(size_t)r * C + c];
    }
}

// ---------------------------------------------------------------------------
// Stage 1/3: tiled bf16 WMMA GEMM.  C[M][N] = A[M][K] @ BT[N][K]^T + bias
// Block tile 128x128, 8 waves, wave tile 32x64 (2x4 WMMA accumulators).
// Double-buffered async global->LDS prefetch overlaps HBM with WMMA.
// MODE 0: fp32 row-major output.  MODE 1: QKV split-head epilogue (bf16),
//         V stored transposed [B*H][HD][S] for the attention PV matmul.
// ---------------------------------------------------------------------------
template<int MODE>
__global__ __launch_bounds__(256) void gemm_bf16_wmma(
    const __bf16* __restrict__ A,
    const __bf16* __restrict__ BT,
    const float*  __restrict__ bias,
    float*   __restrict__ outF,
    __bf16*  __restrict__ Qg,
    __bf16*  __restrict__ Kg,
    __bf16*  __restrict__ Vtg,
    int M, int N, int K)
{
    __shared__ __bf16 sA[2][128][40];   // +8 halves pad: bank-conflict relief
    __shared__ __bf16 sB[2][128][40];

    const int tid    = threadIdx.x;
    const int lane   = tid & 31;
    const int wid    = tid >> 5;
    const int wm     = wid & 3;            // 32-row band
    const int wn     = wid >> 2;           // 64-col band
    const int fr     = lane & 15;
    const int koff   = (lane >> 4) * 8;    // CDNA5 16-bit A/B half-wave K split
    const int blockN = blockIdx.x * 128;
    const int blockM = blockIdx.y * 128;

    // per-thread async prefetch of one 128x32 A tile + one 128x32 B tile
    auto prefetch = [&](int bufi, int k0) {
        int idx = tid;
#pragma unroll
        for (int it = 0; it < 2; ++it, idx += 256) {
            int row = idx >> 2;
            int seg = (idx & 3) * 8;
            async_copy16(&sA[bufi][row][seg], &A[(size_t)(blockM + row) * K + k0 + seg]);
            async_copy16(&sB[bufi][row][seg], &BT[(size_t)(blockN + row) * K + k0 + seg]);
        }
    };

    v8f acc[2][4] = {};

    const int KT = K >> 5;
    prefetch(0, 0);
    int buf = 0;
    for (int kt = 0; kt < KT; ++kt) {
        if (kt + 1 < KT) {
            prefetch(buf ^ 1, (kt + 1) << 5);
            wait_async_le4();   // previous batch (current tile) has landed
        } else {
            wait_async_le0();
        }
        __syncthreads();        // publish tile to all waves

        Frag af[2], bfr[4];
#pragma unroll
        for (int mi = 0; mi < 2; ++mi) {
            const __bf16* p = &sA[buf][wm * 32 + mi * 16 + fr][koff];
            af[mi].u[0] = *reinterpret_cast<const U4*>(p);
            af[mi].u[1] = *reinterpret_cast<const U4*>(p + 16);
        }
#pragma unroll
        for (int ni = 0; ni < 4; ++ni) {
            const __bf16* p = &sB[buf][wn * 64 + ni * 16 + fr][koff];
            bfr[ni].u[0] = *reinterpret_cast<const U4*>(p);
            bfr[ni].u[1] = *reinterpret_cast<const U4*>(p + 16);
        }
#pragma unroll
        for (int mi = 0; mi < 2; ++mi)
#pragma unroll
            for (int ni = 0; ni < 4; ++ni)
                acc[mi][ni] = WMMA_BF16(af[mi].v, bfr[ni].v, acc[mi][ni]);

        __syncthreads();        // all reads done before buf is overwritten
        buf ^= 1;
    }

    // Epilogue: C/D layout = VGPR i -> row i (lanes 0-15) / row 8+i (lanes 16-31)
    const int cn   = lane & 15;
    const int rsel = lane >> 4;
#pragma unroll
    for (int mi = 0; mi < 2; ++mi) {
#pragma unroll
        for (int ni = 0; ni < 4; ++ni) {
            const int col = blockN + wn * 64 + ni * 16 + cn;
            const float bv = bias[col];
#pragma unroll
            for (int i = 0; i < 8; ++i) {
                const int row = blockM + wm * 32 + mi * 16 + rsel * 8 + i;
                float v = acc[mi][ni][i] + bv;
                if (MODE == 0) {
                    outF[(size_t)row * N + col] = v;
                } else {
                    const int which = col >> 11;   // 0:Q 1:K 2:V (N = 3*2048)
                    const int cc = col & 2047;
                    const int h  = cc >> 7;        // head
                    const int d  = cc & 127;       // dim in head
                    const int b  = row >> 11;
                    const int s  = row & 2047;
                    const size_t bh = (size_t)(b * 16 + h);
                    const __bf16 q = (__bf16)v;
                    if (which == 0)      Qg [(bh * 2048 + s) * 128 + d]  = q;
                    else if (which == 1) Kg [(bh * 2048 + s) * 128 + d]  = q;
                    else                 Vtg[(bh * 128  + d) * 2048 + s] = q;  // transposed
                }
            }
        }
    }
}

// ---------------------------------------------------------------------------
// Stage 2: fused causal attention, flash-style online softmax.
// grid = (S/128, B*H); 8 waves/block; wave owns 16 query rows, HD=128.
// Double-buffered async K/V^T tiles; scores use two independent WMMA chains.
// ---------------------------------------------------------------------------
__global__ __launch_bounds__(256) void attn_fused_wmma(
    const __bf16* __restrict__ Qg,
    const __bf16* __restrict__ Kg,
    const __bf16* __restrict__ Vtg,
    __bf16* __restrict__ Oout)
{
    __shared__ __bf16 sK[2][32][136];   // [key][hd]   (+pad)
    __shared__ __bf16 sV[2][128][40];   // [hd][key]   (+pad) — transposed in global
    __shared__ __bf16 sP[8][16][32];    // per-wave P staging (C-layout -> A-layout)

    const int tid  = threadIdx.x;
    const int lane = tid & 31;
    const int w    = tid >> 5;
    const int fr   = lane & 15;
    const int koff = (lane >> 4) * 8;
    const int cn   = fr;
    const int rsel = lane >> 4;

    const int bh    = blockIdx.y;
    const int qbase = blockIdx.x * 128;
    const int qw    = qbase + w * 16;

    auto prefetch = [&](int bufi, int kb) {
        int idx = tid;
#pragma unroll
        for (int it = 0; it < 2; ++it, idx += 256) {
            int row = idx >> 4;
            int seg = (idx & 15) * 8;
            async_copy16(&sK[bufi][row][seg],
                         &Kg[((size_t)bh * 2048 + kb + row) * 128 + seg]);
        }
        idx = tid;
#pragma unroll
        for (int it = 0; it < 2; ++it, idx += 256) {
            int row = idx >> 2;
            int seg = (idx & 3) * 8;
            async_copy16(&sV[bufi][row][seg],
                         &Vtg[((size_t)bh * 128 + row) * 2048 + kb + seg]);
        }
    };

    // Q fragments for this wave's 16 rows (4 chunks over HD=128), kept resident
    Frag qf[4];
    {
        const __bf16* qrow = &Qg[((size_t)bh * 2048 + (qw + fr)) * 128];
#pragma unroll
        for (int c = 0; c < 4; ++c) {
            qf[c].u[0] = *reinterpret_cast<const U4*>(qrow + c * 32 + koff);
            qf[c].u[1] = *reinterpret_cast<const U4*>(qrow + c * 32 + koff + 16);
        }
    }

    v8f oacc[8] = {};
    float mrow[8], lrow[8];
#pragma unroll
    for (int i = 0; i < 8; ++i) { mrow[i] = -1.0e30f; lrow[i] = 0.0f; }

    const float scale = 0.08838834764831845f;   // 1/sqrt(128)
    const int nkt = blockIdx.x * 4 + 4;         // 32-key tiles covering keys < qbase+128

    prefetch(0, 0);
    int buf = 0;
    for (int jt = 0; jt < nkt; ++jt) {
        const int kb = jt * 32;
        if (jt + 1 < nkt) {
            prefetch(buf ^ 1, kb + 32);
            wait_async_le4();
        } else {
            wait_async_le0();
        }
        __syncthreads();

        if (kb <= qw + 15) {   // wave-uniform causal skip
            // scores: two independent 16x16 WMMA chains over 4 HD chunks
            v8f sc0 = {}, sc1 = {};
#pragma unroll
            for (int c = 0; c < 4; ++c) {
                Frag kf0, kf1;
                const __bf16* p0 = &sK[buf][fr][c * 32 + koff];
                const __bf16* p1 = &sK[buf][16 + fr][c * 32 + koff];
                kf0.u[0] = *reinterpret_cast<const U4*>(p0);
                kf0.u[1] = *reinterpret_cast<const U4*>(p0 + 16);
                kf1.u[0] = *reinterpret_cast<const U4*>(p1);
                kf1.u[1] = *reinterpret_cast<const U4*>(p1 + 16);
                sc0 = WMMA_BF16(qf[c].v, kf0.v, sc0);
                sc1 = WMMA_BF16(qf[c].v, kf1.v, sc1);
            }

            // mask + scale, online softmax (rows live in 16-lane half-waves)
            float pvv[2][8];
            float mnew[8];
#pragma unroll
            for (int i = 0; i < 8; ++i) {
                const int q = qw + rsel * 8 + i;
                float v0 = sc0[i];
                float v1 = sc1[i];
                v0 = ((kb + cn)      <= q) ? v0 * scale : -10000.0f;
                v1 = ((kb + 16 + cn) <= q) ? v1 * scale : -10000.0f;
                pvv[0][i] = v0; pvv[1][i] = v1;
                float t = fmaxf(v0, v1);
#pragma unroll
                for (int off = 8; off > 0; off >>= 1)
                    t = fmaxf(t, __shfl_xor(t, off, 32));
                mnew[i] = fmaxf(mrow[i], t);
            }

#pragma unroll
            for (int i = 0; i < 8; ++i) {
                const float mn    = mnew[i];
                const float alpha = __expf(mrow[i] - mn);
                float p0 = __expf(pvv[0][i] - mn);
                float p1 = __expf(pvv[1][i] - mn);
                sP[w][rsel * 8 + i][cn]      = (__bf16)p0;
                sP[w][rsel * 8 + i][16 + cn] = (__bf16)p1;
                float rs = p0 + p1;
#pragma unroll
                for (int off = 8; off > 0; off >>= 1)
                    rs += __shfl_xor(rs, off, 32);
                lrow[i] = lrow[i] * alpha + rs;
                mrow[i] = mn;
#pragma unroll
                for (int t2 = 0; t2 < 8; ++t2)
                    oacc[t2][i] *= alpha;
            }

            // wave-private LDS bounce: drain DS counter before cross-lane read
            asm volatile("s_wait_dscnt 0x0" ::: "memory");

            Frag pf;
            {
                const __bf16* p = &sP[w][fr][koff];
                pf.u[0] = *reinterpret_cast<const U4*>(p);
                pf.u[1] = *reinterpret_cast<const U4*>(p + 16);
            }
            // PV: 8 output tiles across HD=128, contraction over 32 keys
#pragma unroll
            for (int t2 = 0; t2 < 8; ++t2) {
                Frag vf;
                const __bf16* p = &sV[buf][t2 * 16 + fr][koff];
                vf.u[0] = *reinterpret_cast<const U4*>(p);
                vf.u[1] = *reinterpret_cast<const U4*>(p + 16);
                oacc[t2] = WMMA_BF16(pf.v, vf.v, oacc[t2]);
            }
        }
        __syncthreads();
        buf ^= 1;
    }

    // normalize + merge heads into bf16 activation [B*S][D] for the proj GEMM
    const int b = bh >> 4;
    const int h = bh & 15;
    float inv[8];
#pragma unroll
    for (int i = 0; i < 8; ++i) inv[i] = 1.0f / lrow[i];
#pragma unroll
    for (int t2 = 0; t2 < 8; ++t2) {
        const int dcol = h * 128 + t2 * 16 + cn;
#pragma unroll
        for (int i = 0; i < 8; ++i) {
            const int q = qw + rsel * 8 + i;
            Oout[((size_t)(b * 2048 + q)) * 2048 + dcol] = (__bf16)(oacc[t2][i] * inv[i]);
        }
    }
}

// ---------------------------------------------------------------------------
// Launcher
// ---------------------------------------------------------------------------
extern "C" void kernel_launch(void* const* d_in, const int* in_sizes, int n_in,
                              void* d_out, int out_size, void* d_ws, size_t ws_size,
                              hipStream_t stream) {
    (void)in_sizes; (void)n_in; (void)out_size; (void)ws_size;

    const float* hidden   = (const float*)d_in[0];
    const float* c_attn_w = (const float*)d_in[1];
    const float* c_attn_b = (const float*)d_in[2];
    const float* c_proj_w = (const float*)d_in[3];
    const float* c_proj_b = (const float*)d_in[4];
    float* out = (float*)d_out;

    const int Bsz = 2, S = 2048, D = 2048;
    const size_t nHid  = (size_t)Bsz * S * D;   // 8388608
    const size_t nWqkv = (size_t)D * 3 * D;     // 12582912
    const size_t nWp   = (size_t)D * D;         // 4194304

    char* w = (char*)d_ws;
    __bf16* hidB  = (__bf16*)w;  w += nHid  * 2;
    __bf16* wqkvT = (__bf16*)w;  w += nWqkv * 2;
    __bf16* wpT   = (__bf16*)w;  w += nWp   * 2;
    __bf16* Qg    = (__bf16*)w;  w += nHid  * 2;
    __bf16* Kg    = (__bf16*)w;  w += nHid  * 2;
    __bf16* Vtg   = (__bf16*)w;  w += nHid  * 2;
    __bf16* attnB = (__bf16*)w;  w += nHid  * 2;   // total 112 MiB

    cvt_bf16_kernel<<<(int)((nHid + 255) / 256), 256, 0, stream>>>(hidden, hidB, (int)nHid);
    cvt_transpose_bf16_kernel<<<(int)((nWqkv + 255) / 256), 256, 0, stream>>>(c_attn_w, wqkvT, D, 3 * D);
    cvt_transpose_bf16_kernel<<<(int)((nWp + 255) / 256), 256, 0, stream>>>(c_proj_w, wpT, D, D);

    // QKV projection + split-heads (V transposed)
    gemm_bf16_wmma<1><<<dim3(6144 / 128, 4096 / 128), 256, 0, stream>>>(
        hidB, wqkvT, c_attn_b, nullptr, Qg, Kg, Vtg, 4096, 6144, 2048);

    // fused causal attention
    attn_fused_wmma<<<dim3(2048 / 128, Bsz * 16), 256, 0, stream>>>(Qg, Kg, Vtg, attnB);

    // output projection (fp32 out + bias)
    gemm_bf16_wmma<0><<<dim3(2048 / 128, 4096 / 128), 256, 0, stream>>>(
        attnB, wpT, c_proj_b, out, nullptr, nullptr, nullptr, 4096, 2048, 2048);
}
